// GraphEncoder_26465588478288
// MI455X (gfx1250) — matrix operand order repaired and checked by reading, not compiled, not run on hardware
//
#include <hip/hip_runtime.h>
#include <hip/hip_bf16.h>
#include <math.h>

// ---------------- problem constants (match reference) ----------------
#define NA    10000
#define NB    20000
#define NFRAG 3000
#define NFE   12000
#define BATCH 256
#define HID   64
#define EPS   1e-5f
#define NEG_INF (-__builtin_huge_valf())

typedef __attribute__((ext_vector_type(16))) _Float16 v16h;
typedef __attribute__((ext_vector_type(8)))  _Float16 v8h;
typedef __attribute__((ext_vector_type(8)))  float    v8f;
typedef __attribute__((ext_vector_type(4)))  float    v4f;

static __host__ __device__ inline int rup(int x, int m) { return (x + m - 1) / m * m; }

// ---------------- f32 -> padded f16 pack (rows -> mult16, K -> mult32) ----------------
__global__ void k_packf16(const float* __restrict__ X, _Float16* __restrict__ Xh,
                          int N, int K, int Np, int Kp) {
  const int tid = blockIdx.x * blockDim.x + threadIdx.x;
  if (tid >= Np * Kp) return;
  const int r = tid / Kp, k = tid - r * Kp;
  Xh[tid] = (_Float16)((r < N && k < K) ? X[(size_t)r * K + k] : 0.f);
}

// ---------------- branch-free WMMA fragment loads from packed f16 ----------------
// A (16x32 f16, contiguous k order, same as B): lane l=row m + group g; needs row halfs
//   [k0+16g .. k0+16g+15]  -> one contiguous 32B load.
static __device__ __forceinline__ v16h loadA(const _Float16* __restrict__ arow, int k0) {
  const int g = (threadIdx.x >> 4) & 1;
  return *(const v16h*)(arow + k0 + 16 * g);
}
// B (32x16 f16 from row-major W[M,Kp] transposed): lane needs halfs
//   [k0+16g .. k0+16g+15] of row col0+n -> one contiguous 32B load.
static __device__ __forceinline__ v16h loadB(const _Float16* __restrict__ brow, int k0) {
  const int g = (threadIdx.x >> 4) & 1;
  return *(const v16h*)(brow + k0 + 16 * g);
}

// ---------------- generic GEMM: Y[N,M] = act(Xh[Np,Kp] @ Wh[Mp,Kp]^T + bias) ----------
__global__ __launch_bounds__(256) void k_gemm(const _Float16* __restrict__ Xh,
                                              const _Float16* __restrict__ Wh,
                                              const float* __restrict__ bias,
                                              float* __restrict__ Y,
                                              int N, int Kp, int M, int relu) {
  const int tilesM = (M + 15) >> 4;
  const int tilesN = (N + 15) >> 4;
  const int id = blockIdx.x * 8 + (threadIdx.x >> 5);
  if (id >= tilesM * tilesN) return;
  const int tm = id % tilesM, tn = id / tilesM;
  const int row0 = tn << 4, col0 = tm << 4;
  const int l = threadIdx.x & 31;
  const int n16 = l & 15;
  const _Float16* arow = Xh + (size_t)(row0 + n16) * Kp;
  const _Float16* brow = Wh + (size_t)(col0 + n16) * Kp;
  v8f acc = {};
  for (int k0 = 0; k0 < Kp; k0 += 32) {
    v16h a = loadA(arow, k0);
    v16h b = loadB(brow, k0);
    acc = __builtin_amdgcn_wmma_f32_16x16x32_f16(false, a, false, b, (short)0, acc,
                                                 false, false);
  }
  const int col = col0 + n16;
  const int rb  = row0 + ((l >> 4) << 3);
#pragma unroll
  for (int d = 0; d < 8; ++d) {
    const int r = rb + d;
    if (r < N && col < M) {
      float y = acc[d] + (bias ? bias[col] : 0.f);
      if (relu) y = fmaxf(y, 0.f);
      Y[(size_t)r * M + col] = y;
    }
  }
}

// ---------------- BatchNorm (training stats) ----------------
__global__ void k_colstats(const float* __restrict__ Y, int N,
                           float* __restrict__ mu, float* __restrict__ rsig) {
  __shared__ float ss[4][64], ss2[4][64];
  const int d = threadIdx.x & 63, q = threadIdx.x >> 6;
  float s = 0.f, s2 = 0.f;
  for (int n = q; n < N; n += 4) {
    const float y = Y[(size_t)n * HID + d];
    s += y; s2 += y * y;
  }
  ss[q][d] = s; ss2[q][d] = s2;
  __syncthreads();
  if (q == 0) {
#pragma unroll
    for (int t = 1; t < 4; ++t) { s += ss[t][d]; s2 += ss2[t][d]; }
    const float m   = s / (float)N;
    const float var = s2 / (float)N - m * m;
    mu[d]   = m;
    rsig[d] = rsqrtf(var + EPS);
  }
}

__global__ void k_bnrelu(float* __restrict__ Y, const float* __restrict__ mu,
                         const float* __restrict__ rsig, const float* __restrict__ g,
                         const float* __restrict__ be, int N) {
  const int tid = blockIdx.x * blockDim.x + threadIdx.x;
  if (tid >= N * HID) return;
  const int d = tid & 63;
  const float y = (Y[tid] - mu[d]) * rsig[d] * g[d] + be[d];
  Y[tid] = fmaxf(y, 0.f);
}

// ---------------- pack conv weights W[4096,C] into per-lane WMMA-B f16 fragments (k = 16g + s) --------
// chunk j covers flattened columns 16j..16j+15 of the [64,64] per-edge weight.
__global__ void k_packB(const float* __restrict__ W, _Float16* __restrict__ P, int C) {
  const int tid = blockIdx.x * blockDim.x + threadIdx.x;  // 256 chunks * 32 lanes
  if (tid >= 256 * 32) return;
  const int j = tid >> 5, l = tid & 31;
  const int n = l & 15, g = l >> 4;
#pragma unroll
  for (int v = 0; v < 8; ++v) {
    const int k = 16 * g + 2 * v;
    float x0 = (k     < C) ? W[(size_t)(j * 16 + n) * C + k]     : 0.f;
    float x1 = (k + 1 < C) ? W[(size_t)(j * 16 + n) * C + k + 1] : 0.f;
    P[((size_t)j * 32 + l) * 16 + 2 * v]     = (_Float16)x0;
    P[((size_t)j * 32 + l) * 16 + 2 * v + 1] = (_Float16)x1;
  }
}

// ---------------- fused NNConv ----------------
// msg[e,o] = sum_h x_src[e][h] * (ef[e] . W[h*64+o,:] + b[h*64+o]); scatter-add to dst.
// One wave per 16-edge tile; Wm chunks exist only in WMMA accumulators.
__global__ __launch_bounds__(256) void k_nnconv(const float* __restrict__ nf,
                                                const _Float16* __restrict__ efh,  // [Ep,32]
                                                const int* __restrict__ src,
                                                const int* __restrict__ dst,
                                                const _Float16* __restrict__ P,
                                                const float* __restrict__ bias,
                                                float* __restrict__ out,
                                                int E) {
  __shared__ float xs[8][16 * HID];  // transposed: xs[wid][h*16+m]
  __shared__ float bsh[4096];
  const int wid = threadIdx.x >> 5;
  const int l   = threadIdx.x & 31;
  const int t   = blockIdx.x * 8 + wid;

  for (int idx = threadIdx.x; idx < 4096; idx += 256) bsh[idx] = bias[idx];
  for (int idx = l; idx < 16 * HID; idx += 32) {
    const int m = idx & 15, h = idx >> 4;
    const int e = t * 16 + m;
    xs[wid][h * 16 + m] = (e < E) ? nf[(size_t)src[e] * HID + h] : 0.f;
  }
  __syncthreads();

  // A fragment from padded edge features: branch-free contiguous loads.
  const int m0 = l & 15, g = l >> 4;
  const _Float16* arow = efh + (size_t)(t * 16 + m0) * 32;
  const v16h a = loadA(arow, 0);

  for (int oc = 0; oc < 4; ++oc) {
    v8f acc = {};
    for (int h = 0; h < HID; ++h) {
      const int j = h * 4 + oc;  // 0..255
      const v16h b = *(const v16h*)(P + ((size_t)j * 32 + l) * 16);
      const int jn = (j + 4 < 256) ? j + 4 : j;
      __builtin_prefetch(P + ((size_t)jn * 32 + l) * 16, 0, 1);  // global_prefetch_b8
      v8f c = {};
      c = __builtin_amdgcn_wmma_f32_16x16x32_f16(false, a, false, b, (short)0, c,
                                                 false, false);
      const float bc = bsh[j * 16 + m0];
      const v4f x0 = *(const v4f*)(&xs[wid][h * 16 + 8 * g]);
      const v4f x1 = *(const v4f*)(&xs[wid][h * 16 + 8 * g + 4]);
#pragma unroll
      for (int d = 0; d < 4; ++d) {
        acc[d]     += x0[d] * (c[d]     + bc);
        acc[d + 4] += x1[d] * (c[d + 4] + bc);
      }
    }
    const int o = oc * 16 + m0;
#pragma unroll
    for (int d = 0; d < 8; ++d) {
      const int e = t * 16 + d + 8 * g;
      if (e < E) atomicAdd(&out[(size_t)dst[e] * HID + o], acc[d]);
    }
  }
}

// ---------------- small elementwise / segment kernels ----------------
__global__ void k_fill(float* __restrict__ p, size_t n, float val) {
  const size_t i = (size_t)blockIdx.x * blockDim.x + threadIdx.x;
  if (i < n) p[i] = val;
}

__global__ void k_segsum_a2f(const float* __restrict__ uam, const int* __restrict__ d,
                             float* __restrict__ aug) {
  const int tid = blockIdx.x * blockDim.x + threadIdx.x;
  if (tid >= NA * HID) return;
  const int n = tid >> 6, c = tid & 63;
  atomicAdd(&aug[(size_t)d[n] * HID + c], uam[tid]);
}

__global__ void k_attn(const float* __restrict__ Q, const float* __restrict__ Kf,
                       const float* __restrict__ Kc, const float* __restrict__ Vf,
                       const float* __restrict__ Vc, float* __restrict__ AO) {
  __shared__ float pf[HID], pc[HID], w0[4], w1[4];
  const int n = blockIdx.x, d = threadIdx.x;
  const float q = Q[n * HID + d];
  pf[d] = q * Kf[n * HID + d];
  pc[d] = q * Kc[n * HID + d];
  __syncthreads();
  if (d < 4) {
    float sa = 0.f, sb = 0.f;
    for (int i = 0; i < 16; ++i) { sa += pf[d * 16 + i]; sb += pc[d * 16 + i]; }
    sa *= (1.f / 16.f);  // logits / DK
    sb *= (1.f / 16.f);
    const float mx = fmaxf(sa, sb);
    const float e0 = expf(sa - mx), e1 = expf(sb - mx);
    const float inv = 1.f / (e0 + e1);
    w0[d] = e0 * inv; w1[d] = e1 * inv;
  }
  __syncthreads();
  const int h = d >> 4;
  AO[n * HID + d] = w0[h] * Vf[n * HID + d] + w1[h] * Vc[n * HID + d];
}

__global__ void k_gru(const float* __restrict__ gi, const float* __restrict__ gh,
                      float* __restrict__ h, int N) {
  const int tid = blockIdx.x * blockDim.x + threadIdx.x;
  if (tid >= N * HID) return;
  const int n = tid >> 6, d = tid & 63;
  const float* gin = gi + (size_t)n * 192;
  const float* ghn = gh + (size_t)n * 192;
  const float r = 1.f / (1.f + expf(-(gin[d] + ghn[d])));
  const float z = 1.f / (1.f + expf(-(gin[64 + d] + ghn[64 + d])));
  const float gg = tanhf(gin[128 + d] + r * ghn[128 + d]);
  h[tid] = (1.f - z) * gg + z * h[tid];
}

static __device__ __forceinline__ void atomicMaxF(float* addr, float val) {
  if (val >= 0.f) atomicMax((int*)addr, __float_as_int(val));
  else            atomicMin((unsigned int*)addr, __float_as_uint(val));
}

__global__ void k_nw(const float* __restrict__ nf, const float* __restrict__ Wr,
                     const float* __restrict__ br, float* __restrict__ nw, int N) {
  const int n = blockIdx.x * blockDim.x + threadIdx.x;
  if (n >= N) return;
  float s = br[0];
  for (int d = 0; d < HID; ++d) s += nf[(size_t)n * HID + d] * Wr[d];
  nw[n] = s;
}

__global__ void k_segmax_nw(const float* __restrict__ nw, const int* __restrict__ batch,
                            float* __restrict__ m, int N) {
  const int n = blockIdx.x * blockDim.x + threadIdx.x;
  if (n >= N) return;
  atomicMaxF(&m[batch[n]], nw[n]);
}

__global__ void k_expsum(const float* __restrict__ nw, const int* __restrict__ batch,
                         const float* __restrict__ m, float* __restrict__ s, int N) {
  const int n = blockIdx.x * blockDim.x + threadIdx.x;
  if (n >= N) return;
  const int bb = batch[n];
  atomicAdd(&s[bb], expf(nw[n] - m[bb]));
}

__global__ void k_waccum(const float* __restrict__ nf, const float* __restrict__ nw,
                         const int* __restrict__ batch, const float* __restrict__ m,
                         float* __restrict__ wraw, float* __restrict__ mx, int N) {
  const int tid = blockIdx.x * blockDim.x + threadIdx.x;
  if (tid >= N * HID) return;
  const int n = tid >> 6, d = tid & 63;
  const int bb = batch[n];
  const float e = expf(nw[n] - m[bb]);
  atomicAdd(&wraw[(size_t)bb * HID + d], e * nf[tid]);
  atomicMaxF(&mx[(size_t)bb * HID + d], nf[tid]);
}

__global__ void k_hgraph(const float* __restrict__ wrA, const float* __restrict__ sA,
                         const float* __restrict__ mxA, const float* __restrict__ wrF,
                         const float* __restrict__ sF, const float* __restrict__ mxF,
                         float* __restrict__ hg) {
  const int tid = blockIdx.x * blockDim.x + threadIdx.x;
  if (tid >= BATCH * 256) return;
  const int b = tid >> 8, d = tid & 255;
  float v;
  if      (d < 64)  v = wrA[b * 64 + d]         / sA[b];
  else if (d < 128) v = mxA[b * 64 + (d - 64)];
  else if (d < 192) v = wrF[b * 64 + (d - 128)] / sF[b];
  else              v = mxF[b * 64 + (d - 192)];
  hg[tid] = v;
}

__global__ void k_final(const float* __restrict__ h1, const float* __restrict__ W2,
                        const float* __restrict__ b2, const float* __restrict__ cqW,
                        const float* __restrict__ cqb, float* __restrict__ out) {
  const int b = blockIdx.x * blockDim.x + threadIdx.x;
  if (b >= BATCH) return;
  float o0 = b2[0], o1 = b2[1];
  for (int d = 0; d < 600; ++d) {
    const float h = h1[(size_t)b * 600 + d];
    o0 += h * W2[d];
    o1 += h * W2[600 + d];
  }
  out[b * 2 + 0] = cqW[0] * o0 + cqW[1] * o1 + cqb[0];
  out[b * 2 + 1] = cqW[2] * o0 + cqW[3] * o1 + cqb[1];
}

// ---------------- host orchestration ----------------
static inline void packf16(hipStream_t s, const float* X, _Float16* Xh,
                           int N, int K, int Np, int Kp) {
  const int total = Np * Kp;
  k_packf16<<<(total + 255) / 256, 256, 0, s>>>(X, Xh, N, K, Np, Kp);
}
static inline void gemm_f(hipStream_t s, const float* X, _Float16* Xp,
                          const _Float16* Wh, const float* bias, float* Y,
                          int N, int K, int Kp, int M, int relu) {
  packf16(s, X, Xp, N, K, rup(N, 16), Kp);
  const int tiles = ((M + 15) >> 4) * ((N + 15) >> 4);
  k_gemm<<<(tiles + 7) / 8, 256, 0, s>>>(Xp, Wh, bias, Y, N, Kp, M, relu);
}
static inline void fill(hipStream_t s, float* p, size_t n, float v) {
  k_fill<<<(int)((n + 255) / 256), 256, 0, s>>>(p, n, v);
}

extern "C" void kernel_launch(void* const* d_in, const int* in_sizes, int n_in,
                              void* d_out, int out_size, void* d_ws, size_t ws_size,
                              hipStream_t stream) {
  // ---- inputs (setup_inputs dict order; params dict flattened in insertion order) ----
  const float* af   = (const float*)d_in[0];
  const float* bf   = (const float*)d_in[1];
  const float* fnf  = (const float*)d_in[2];
  const float* fef  = (const float*)d_in[3];
  const int* a_src  = (const int*)d_in[4];
  const int* a_dst  = (const int*)d_in[5];
  const int* f_src  = (const int*)d_in[6];
  const int* f_dst  = (const int*)d_in[7];
  const int* a2f    = (const int*)d_in[8];
  const int* abat   = (const int*)d_in[9];
  const int* fbat   = (const int*)d_in[10];
  int i = 11;
  const float* ae_W1 = (const float*)d_in[i++]; const float* ae_b1 = (const float*)d_in[i++];
  const float* ae_g1 = (const float*)d_in[i++]; const float* ae_be1 = (const float*)d_in[i++];
  const float* ae_W2 = (const float*)d_in[i++]; const float* ae_b2 = (const float*)d_in[i++];
  const float* ae_g2 = (const float*)d_in[i++]; const float* ae_be2 = (const float*)d_in[i++];
  const float* me_W1 = (const float*)d_in[i++]; const float* me_b1 = (const float*)d_in[i++];
  const float* me_g1 = (const float*)d_in[i++]; const float* me_be1 = (const float*)d_in[i++];
  const float* me_W2 = (const float*)d_in[i++]; const float* me_b2 = (const float*)d_in[i++];
  const float* me_g2 = (const float*)d_in[i++]; const float* me_be2 = (const float*)d_in[i++];
  const float* aef_W = (const float*)d_in[i++]; const float* aef_b = (const float*)d_in[i++];
  const float* mef_W = (const float*)d_in[i++]; const float* mef_b = (const float*)d_in[i++];
  const float* la_Wq = (const float*)d_in[i++]; const float* la_Wk = (const float*)d_in[i++];
  const float* la_Wv = (const float*)d_in[i++]; const float* la_Wo = (const float*)d_in[i++];
  const float* la_bo = (const float*)d_in[i++];
  const float* agru_Wih = (const float*)d_in[i++]; const float* agru_Whh = (const float*)d_in[i++];
  const float* agru_bih = (const float*)d_in[i++]; const float* agru_bhh = (const float*)d_in[i++];
  const float* mgru_Wih = (const float*)d_in[i++]; const float* mgru_Whh = (const float*)d_in[i++];
  const float* mgru_bih = (const float*)d_in[i++]; const float* mgru_bhh = (const float*)d_in[i++];
  const float* ar_W = (const float*)d_in[i++]; const float* ar_b = (const float*)d_in[i++];
  const float* mr_W = (const float*)d_in[i++]; const float* mr_b = (const float*)d_in[i++];
  const float* spu_W1 = (const float*)d_in[i++]; const float* spu_b1 = (const float*)d_in[i++];
  const float* spu_W2 = (const float*)d_in[i++]; const float* spu_b2 = (const float*)d_in[i++];
  const float* cq_W = (const float*)d_in[i++]; const float* cq_b = (const float*)d_in[i++];

  // padded edge-row counts (cover every launched wave's 16-edge tile)
  const int blocksA = (NB / 16 + 7) / 8,  EpA = blocksA * 128;   // 20096
  const int blocksM = (NFE / 16 + 7) / 8, EpM = blocksM * 128;   // 12032

  // ---- workspace carve-up (~35 MB total) ----
  char* w = (char*)d_ws;
  auto alloc = [&](size_t bytes) -> void* {
    void* p = (void*)w;
    w += (bytes + 255) & ~(size_t)255;
    return p;
  };
  float* t1    = (float*)alloc((size_t)NA * HID * 4);      // encoder layer-1 scratch
  float* uaf   = (float*)alloc((size_t)NA * HID * 4);
  float* ufnf  = (float*)alloc((size_t)NFRAG * HID * 4);
  float* uam   = (float*)alloc((size_t)NA * HID * 4);
  float* ufnm  = (float*)alloc((size_t)NFRAG * HID * 4);
  float* aug   = (float*)alloc((size_t)NFRAG * HID * 4);
  float* Qb    = (float*)alloc((size_t)NFRAG * HID * 4);
  float* Kfb   = (float*)alloc((size_t)NFRAG * HID * 4);
  float* Kcb   = (float*)alloc((size_t)NFRAG * HID * 4);
  float* Vfb   = (float*)alloc((size_t)NFRAG * HID * 4);
  float* Vcb   = (float*)alloc((size_t)NFRAG * HID * 4);
  float* AOb   = (float*)alloc((size_t)NFRAG * HID * 4);
  float* gi    = (float*)alloc((size_t)NA * 192 * 4);
  float* gh    = (float*)alloc((size_t)NA * 192 * 4);
  _Float16* packA = (_Float16*)alloc((size_t)256 * 32 * 16 * 2);
  _Float16* packM = (_Float16*)alloc((size_t)256 * 32 * 16 * 2);
  _Float16* efhA  = (_Float16*)alloc((size_t)EpA * 32 * 2);
  _Float16* efhM  = (_Float16*)alloc((size_t)EpM * 32 * 2);
  _Float16* Xp    = (_Float16*)alloc((size_t)NA * 64 * 2);     // activation pack scratch
  // pre-packed f16 weights (row-major, K padded to 32)
  _Float16* w_meW1 = (_Float16*)alloc((size_t)64 * 64 * 2);
  _Float16* w_meW2 = (_Float16*)alloc((size_t)64 * 64 * 2);
  _Float16* w_aeW1 = (_Float16*)alloc((size_t)64 * 64 * 2);
  _Float16* w_aeW2 = (_Float16*)alloc((size_t)64 * 64 * 2);
  _Float16* w_q    = (_Float16*)alloc((size_t)64 * 64 * 2);
  _Float16* w_k    = (_Float16*)alloc((size_t)64 * 64 * 2);
  _Float16* w_v    = (_Float16*)alloc((size_t)64 * 64 * 2);
  _Float16* w_o    = (_Float16*)alloc((size_t)64 * 64 * 2);
  _Float16* w_mih  = (_Float16*)alloc((size_t)192 * 64 * 2);
  _Float16* w_mhh  = (_Float16*)alloc((size_t)192 * 64 * 2);
  _Float16* w_aih  = (_Float16*)alloc((size_t)192 * 64 * 2);
  _Float16* w_ahh  = (_Float16*)alloc((size_t)192 * 64 * 2);
  _Float16* w_spu1 = (_Float16*)alloc((size_t)608 * 256 * 2);
  float* mu    = (float*)alloc(64 * 4);
  float* rsig  = (float*)alloc(64 * 4);
  float* nwA   = (float*)alloc((size_t)NA * 4);
  float* nwF   = (float*)alloc((size_t)NFRAG * 4);
  float* mA    = (float*)alloc(BATCH * 4);
  float* sA    = (float*)alloc(BATCH * 4);
  float* wrA   = (float*)alloc((size_t)BATCH * HID * 4);
  float* mxA   = (float*)alloc((size_t)BATCH * HID * 4);
  float* mF    = (float*)alloc(BATCH * 4);
  float* sF    = (float*)alloc(BATCH * 4);
  float* wrF   = (float*)alloc((size_t)BATCH * HID * 4);
  float* mxF   = (float*)alloc((size_t)BATCH * HID * 4);
  float* hg    = (float*)alloc((size_t)BATCH * 256 * 4);
  float* h1    = (float*)alloc((size_t)BATCH * 600 * 4);
  (void)ws_size; (void)n_in; (void)in_sizes; (void)out_size;

  hipStream_t s = stream;

  // ---- pre-pack all GEMM weights + edge features to f16 ----
  packf16(s, me_W1, w_meW1, 64, 50, 64, 64);
  packf16(s, me_W2, w_meW2, 64, 64, 64, 64);
  packf16(s, ae_W1, w_aeW1, 64, 39, 64, 64);
  packf16(s, ae_W2, w_aeW2, 64, 64, 64, 64);
  packf16(s, la_Wq, w_q, 64, 64, 64, 64);
  packf16(s, la_Wk, w_k, 64, 64, 64, 64);
  packf16(s, la_Wv, w_v, 64, 64, 64, 64);
  packf16(s, la_Wo, w_o, 64, 64, 64, 64);
  packf16(s, mgru_Wih, w_mih, 192, 64, 192, 64);
  packf16(s, mgru_Whh, w_mhh, 192, 64, 192, 64);
  packf16(s, agru_Wih, w_aih, 192, 64, 192, 64);
  packf16(s, agru_Whh, w_ahh, 192, 64, 192, 64);
  packf16(s, spu_W1, w_spu1, 600, 256, 608, 256);
  packf16(s, bf,  efhA, NB, 10, EpA, 32);
  packf16(s, fef, efhM, NFE, 14, EpM, 32);
  k_packB<<<32, 256, 0, s>>>(aef_W, packA, 10);
  k_packB<<<32, 256, 0, s>>>(mef_W, packM, 14);

  // ---- encoders (Linear + batch-stat BN + ReLU, two layers each) ----
  gemm_f(s, fnf, Xp, w_meW1, me_b1, t1, NFRAG, 50, 64, HID, 0);
  k_colstats<<<1, 256, 0, s>>>(t1, NFRAG, mu, rsig);
  k_bnrelu<<<(NFRAG * HID + 255) / 256, 256, 0, s>>>(t1, mu, rsig, me_g1, me_be1, NFRAG);
  gemm_f(s, t1, Xp, w_meW2, me_b2, ufnf, NFRAG, HID, 64, HID, 0);
  k_colstats<<<1, 256, 0, s>>>(ufnf, NFRAG, mu, rsig);
  k_bnrelu<<<(NFRAG * HID + 255) / 256, 256, 0, s>>>(ufnf, mu, rsig, me_g2, me_be2, NFRAG);

  gemm_f(s, af, Xp, w_aeW1, ae_b1, t1, NA, 39, 64, HID, 0);
  k_colstats<<<1, 256, 0, s>>>(t1, NA, mu, rsig);
  k_bnrelu<<<(NA * HID + 255) / 256, 256, 0, s>>>(t1, mu, rsig, ae_g1, ae_be1, NA);
  gemm_f(s, t1, Xp, w_aeW2, ae_b2, uaf, NA, HID, 64, HID, 0);
  k_colstats<<<1, 256, 0, s>>>(uaf, NA, mu, rsig);
  k_bnrelu<<<(NA * HID + 255) / 256, 256, 0, s>>>(uaf, mu, rsig, ae_g2, ae_be2, NA);

  // ---- 2 message-passing steps ----
  for (int step = 0; step < 2; ++step) {
    fill(s, ufnm, (size_t)NFRAG * HID, 0.f);
    fill(s, uam,  (size_t)NA * HID, 0.f);
    k_nnconv<<<blocksM, 256, 0, s>>>(ufnf, efhM, f_src, f_dst, packM, mef_b, ufnm, NFE);
    k_nnconv<<<blocksA, 256, 0, s>>>(uaf, efhA, a_src, a_dst, packA, aef_b, uam, NB);
    fill(s, aug, (size_t)NFRAG * HID, 0.f);
    k_segsum_a2f<<<(NA * HID + 255) / 256, 256, 0, s>>>(uam, a2f, aug);

    // local augmentation attention (Q from motif; K/V from [fine=aug, coarse=ufnm])
    gemm_f(s, ufnf, Xp, w_q, nullptr, Qb,  NFRAG, HID, 64, HID, 0);
    gemm_f(s, aug,  Xp, w_k, nullptr, Kfb, NFRAG, HID, 64, HID, 0);
    gemm_f(s, ufnm, Xp, w_k, nullptr, Kcb, NFRAG, HID, 64, HID, 0);
    gemm_f(s, aug,  Xp, w_v, nullptr, Vfb, NFRAG, HID, 64, HID, 0);
    gemm_f(s, ufnm, Xp, w_v, nullptr, Vcb, NFRAG, HID, 64, HID, 0);
    k_attn<<<NFRAG, 64, 0, s>>>(Qb, Kfb, Kcb, Vfb, Vcb, AOb);
    gemm_f(s, AOb, Xp, w_o, la_bo, ufnm, NFRAG, HID, 64, HID, 0);

    // GRUs (motif then atom)
    gemm_f(s, ufnm, Xp, w_mih, mgru_bih, gi, NFRAG, HID, 64, 192, 0);
    gemm_f(s, ufnf, Xp, w_mhh, mgru_bhh, gh, NFRAG, HID, 64, 192, 0);
    k_gru<<<(NFRAG * HID + 255) / 256, 256, 0, s>>>(gi, gh, ufnf, NFRAG);
    gemm_f(s, uam, Xp, w_aih, agru_bih, gi, NA, HID, 64, 192, 0);
    gemm_f(s, uaf, Xp, w_ahh, agru_bhh, gh, NA, HID, 64, 192, 0);
    k_gru<<<(NA * HID + 255) / 256, 256, 0, s>>>(gi, gh, uaf, NA);
  }

  // ---- readouts ----
  k_nw<<<(NA + 255) / 256, 256, 0, s>>>(uaf, ar_W, ar_b, nwA, NA);
  fill(s, mA, BATCH, NEG_INF);
  k_segmax_nw<<<(NA + 255) / 256, 256, 0, s>>>(nwA, abat, mA, NA);
  fill(s, sA, BATCH, 0.f);
  fill(s, wrA, (size_t)BATCH * HID, 0.f);
  fill(s, mxA, (size_t)BATCH * HID, NEG_INF);
  k_expsum<<<(NA + 255) / 256, 256, 0, s>>>(nwA, abat, mA, sA, NA);
  k_waccum<<<(NA * HID + 255) / 256, 256, 0, s>>>(uaf, nwA, abat, mA, wrA, mxA, NA);

  k_nw<<<(NFRAG + 255) / 256, 256, 0, s>>>(ufnf, mr_W, mr_b, nwF, NFRAG);
  fill(s, mF, BATCH, NEG_INF);
  k_segmax_nw<<<(NFRAG + 255) / 256, 256, 0, s>>>(nwF, fbat, mF, NFRAG);
  fill(s, sF, BATCH, 0.f);
  fill(s, wrF, (size_t)BATCH * HID, 0.f);
  fill(s, mxF, (size_t)BATCH * HID, NEG_INF);
  k_expsum<<<(NFRAG + 255) / 256, 256, 0, s>>>(nwF, fbat, mF, sF, NFRAG);
  k_waccum<<<(NFRAG * HID + 255) / 256, 256, 0, s>>>(ufnf, nwF, fbat, mF, wrF, mxF, NFRAG);

  // ---- head: h_graph -> 600 relu -> 2 -> 2 ----
  k_hgraph<<<(BATCH * 256 + 255) / 256, 256, 0, s>>>(wrA, sA, mxA, wrF, sF, mxF, hg);
  gemm_f(s, hg, Xp, w_spu1, spu_b1, h1, BATCH, 256, 256, 600, 1);
  k_final<<<(BATCH + 255) / 256, 256, 0, s>>>(h1, spu_W2, spu_b2, cq_W, cq_b,
                                              (float*)d_out);
}
